// T5Attention_34522947125260
// MI455X (gfx1250) — matrix operand order repaired and checked by reading, not compiled, not run on hardware
//
#include <hip/hip_runtime.h>
#include <hip/hip_bf16.h>
#include <math.h>

// ---------------------------------------------------------------------------
// T5 attention forward for MI455X (gfx1250, wave32, WMMA + TDM).
//   hidden[2,2048,1024] fp32 -> (output[2,2048,1024], bias[1,16,2048,2048])
// Matmuls on v_wmma_f32_16x16x32_bf16 (fp32 downconverted to bf16, fp32
// accumulate). Flash-style online softmax => no SxS score tensor. K tiles are
// DMA'd into LDS by the Tensor Data Mover (tensor_load_to_lds, TENSORcnt)
// with double buffering so the DMA overlaps WMMA work.
// ---------------------------------------------------------------------------

#define SEQ   2048
#define DM    1024
#define HEADS 16
#define DKV   64
#define BATCH 2
#define LUT_N (2*SEQ - 1)   // 4095 relative-position deltas
#define KTILES (SEQ / 32)   // 64 key tiles of 32

typedef __attribute__((ext_vector_type(16))) __bf16 v16bf_t;
typedef __attribute__((ext_vector_type(8)))  float  v8f_t;
typedef __attribute__((ext_vector_type(4)))  unsigned int u32x4_t;
typedef __attribute__((ext_vector_type(8)))  unsigned int u32x8_t;

union BF16Frag {
    v16bf_t v;
    __bf16  e[16];
    uint4   q[2];
};

__device__ __forceinline__ v8f_t wmma_bf16(v16bf_t a, v16bf_t b, v8f_t c) {
    // D(16x16 f32) = A(16x32 bf16) * B(32x16 bf16) + C
    return __builtin_amdgcn_wmma_f32_16x16x32_bf16(
        /*neg_a=*/false, a, /*neg_b=*/false, b,
        /*c_mod=*/(short)0, c, /*reuse_a=*/false, /*reuse_b=*/false);
}

// TDM: DMA one 32(keys) x 64(d) bf16 tile from global into LDS.
// D# per CDNA5 ISA: 2D tensor -> groups 0,1 only.
__device__ __forceinline__ void tdm_load_k_tile(const __bf16* gsrc, void* lds_dst) {
    unsigned long long ga = (unsigned long long)(uintptr_t)gsrc;
    unsigned int       la = (unsigned int)(uintptr_t)lds_dst;   // low 32 = LDS offset
    u32x4_t g0;
    g0[0] = 1u;                                  // count=1, user descriptor
    g0[1] = la;                                  // lds_addr
    g0[2] = (unsigned int)ga;                    // global_addr[31:0]
    g0[3] = (unsigned int)(ga >> 32) | (2u << 30); // global_addr[56:32] | type=2
    u32x8_t g1;
    g1[0] = 1u << 16;                            // data_size=1 (2 bytes)
    g1[1] = 64u << 16;                           // tensor_dim0[15:0]=64 (bits 63:48)
    g1[2] = 32u << 16;                           // tensor_dim1[15:0]=32 (bits 95:80)
    g1[3] = 64u << 16;                           // tile_dim0=64 (bits 127:112)
    g1[4] = 32u;                                 // tile_dim1=32, tile_dim2=0
    g1[5] = 1024u;                               // tensor_dim0_stride[31:0]=1024 elems
    g1[6] = 0u;                                  // stride hi / dim1_stride (unused, 2D)
    g1[7] = 0u;
    asm volatile("tensor_load_to_lds %0, %1" :: "s"(g0), "s"(g1) : "memory");
}

// ---------------------------------------------------------------------------
// Kernel 1: T5 relative-position bucket LUT over delta = k - q in [-2047,2047]
// ---------------------------------------------------------------------------
__global__ void t5_bucket_lut(int* __restrict__ lut) {
    int i = blockIdx.x * blockDim.x + threadIdx.x;
    if (i >= LUT_N) return;
    int delta = i - (SEQ - 1);          // mem - ctx
    int n = -delta;                     // reference: n = -relative_position
    int ret = 0;
    if (n < 0) { ret = 16; n = -n; }    // num_buckets//2 = 16
    int v;
    if (n < 8) {                        // max_exact = 8
        v = n;
    } else {
        float f = logf((float)n / 8.0f) / logf(128.0f / 8.0f) * 8.0f;
        v = 8 + (int)f;
        if (v > 15) v = 15;
    }
    lut[i] = v + ret;
}

// ---------------------------------------------------------------------------
// Kernel 2: materialize position_bias [1,H,S,S] into d_out tail region
// ---------------------------------------------------------------------------
__global__ void __launch_bounds__(256)
t5_bias_out(const int* __restrict__ lut, const float* __restrict__ rel,
            float* __restrict__ out) {
    int hq = blockIdx.x;                 // h*SEQ + q
    int h = hq / SEQ, q = hq % SEQ;
    float* row = out + (size_t)hq * SEQ;
    for (int k = threadIdx.x; k < SEQ; k += blockDim.x)
        row[k] = rel[lut[k - q + (SEQ - 1)] * HEADS + h];
}

// ---------------------------------------------------------------------------
// Kernel 3: C_bf16[M,N] = A_f32[M,K] @ B_f32[K,N]   (one 16x64 tile per wave)
// ---------------------------------------------------------------------------
__global__ void __launch_bounds__(128)
gemm_f32_bf16(const float* __restrict__ A, const float* __restrict__ Bm,
              __bf16* __restrict__ C, int M, int N, int K) {
    int lane = threadIdx.x & 31;
    int w    = threadIdx.x >> 5;
    int tile = blockIdx.x * 4 + w;
    int ntn  = N / 64;
    int row0 = (tile / ntn) * 16;
    int col0 = (tile % ntn) * 64;
    if (row0 >= M) return;

    int m    = lane & 15;
    int base = (lane < 16) ? 0 : 8;     // A-frag K mapping (16-bit A 16x32)
    int off  = (lane < 16) ? 0 : 8;     // C/D row mapping

    v8f_t acc[4] = {};
    for (int c = 0; c < K / 32; ++c) {
        BF16Frag a;
        const float* arow = A + (size_t)(row0 + m) * K + c * 32;
#pragma unroll
        for (int j = 0; j < 8; ++j) a.e[j]     = (__bf16)arow[base + j];
#pragma unroll
        for (int j = 0; j < 8; ++j) a.e[8 + j] = (__bf16)arow[16 + base + j];

        const float* brow = Bm + (size_t)(c * 32) * N + col0 + m; // lane <-> N col
#pragma unroll
        for (int t = 0; t < 4; ++t) {
            BF16Frag b;
#pragma unroll
            for (int j = 0; j < 16; ++j) b.e[j] = (__bf16)brow[(size_t)((j & 7) + base + ((j >> 3) << 4)) * N + t * 16];
            acc[t] = wmma_bf16(a.v, b.v, acc[t]);
        }
    }
#pragma unroll
    for (int t = 0; t < 4; ++t)
#pragma unroll
        for (int r = 0; r < 8; ++r)
            C[(size_t)(row0 + off + r) * N + col0 + t * 16 + m] = (__bf16)acc[t][r];
}

// ---------------------------------------------------------------------------
// Kernel 4: flash attention, one (b,h,16-query tile) per wave.
// K tiles double-buffered via TDM; per 32-key step: 4 WMMA scores + 4 WMMA PV.
// ---------------------------------------------------------------------------
__global__ void __launch_bounds__(128)
t5_flash_attn(const __bf16* __restrict__ Q, const __bf16* __restrict__ Kb,
              const __bf16* __restrict__ V, const float* __restrict__ mask,
              const float* __restrict__ rel, const int* __restrict__ lut,
              __bf16* __restrict__ Ctx) {
    __shared__ int    s_lut[LUT_N];                                  // 16 KB
    __shared__ __attribute__((aligned(16))) __bf16 s_k[4][2][32][DKV]; // 32 KB
    __shared__ __attribute__((aligned(16))) __bf16 s_p[4][16][32];     // 4 KB
    __shared__ float  s_bias[4][32];

    int lane = threadIdx.x & 31;
    int w    = threadIdx.x >> 5;
    int wid  = blockIdx.x * 4 + w;          // 0..4095
    int qt   = wid & 127;
    int h    = (wid >> 7) & 15;
    int b    = wid >> 11;

    for (int i = threadIdx.x; i < LUT_N; i += blockDim.x) s_lut[i] = lut[i];
    s_bias[w][lane] = rel[lane * HEADS + h];
    __syncthreads();

    int m    = lane & 15;
    int base = (lane < 16) ? 0 : 8;
    int off  = (lane < 16) ? 0 : 8;

    const __bf16* kbase = Kb + (size_t)(b * SEQ) * DM + h * DKV;

    // Q fragments for the two K=32 chunks of d=64 (contiguous 16B loads)
    BF16Frag aq[2];
#pragma unroll
    for (int c = 0; c < 2; ++c) {
        const __bf16* qrow = Q + (size_t)(b * SEQ + qt * 16 + m) * DM + h * DKV + c * 32;
        aq[c].q[0] = *(const uint4*)(qrow + base);
        aq[c].q[1] = *(const uint4*)(qrow + 16 + base);
    }

    v8f_t acc[4] = {};
    float rmax[8], rsum[8];
#pragma unroll
    for (int r = 0; r < 8; ++r) { rmax[r] = -INFINITY; rsum[r] = 0.0f; }

    // prime the pipeline: DMA K tile 0 into buffer 0
    tdm_load_k_tile(kbase, &s_k[w][0][0][0]);

    for (int kt = 0; kt < KTILES; ++kt) {
        int k0  = kt * 32;
        int buf = kt & 1;
        // overlap: kick off next tile's DMA, then wait for current tile only
        if (kt + 1 < KTILES) {
            tdm_load_k_tile(kbase + (size_t)(k0 + 32) * DM, &s_k[w][buf ^ 1][0][0]);
            __builtin_amdgcn_s_wait_tensorcnt(1);   // in-order => current tile done
        } else {
            __builtin_amdgcn_s_wait_tensorcnt(0);
        }
        // prefetch next V rows while we compute (global_prefetch_b8)
        if (kt + 1 < KTILES)
            __builtin_prefetch(V + (size_t)(b * SEQ + k0 + 32 + lane) * DM + h * DKV, 0, 0);

        // scores: S(16x32) = Q(16x64) * K^T ; B-frag lane <-> key col, elements over d
        v8f_t sc[2];
#pragma unroll
        for (int t = 0; t < 2; ++t) {
            BF16Frag bk0, bk1;
#pragma unroll
            for (int j = 0; j < 16; ++j) {
                bk0.e[j] = s_k[w][buf][t * 16 + m][(j & 7) + base + ((j >> 3) << 4)];        // chunk d = 0..31
                bk1.e[j] = s_k[w][buf][t * 16 + m][32 + (j & 7) + base + ((j >> 3) << 4)];   // chunk d = 32..63
            }
            v8f_t c0 = {};
            c0 = wmma_bf16(aq[0].v, bk0.v, c0);
            c0 = wmma_bf16(aq[1].v, bk1.v, c0);
            sc[t] = c0;
        }

        // T5 bias + mask, then online softmax over this 32-key chunk
#pragma unroll
        for (int t = 0; t < 2; ++t) {
            int ng = k0 + t * 16 + m;                       // key index (N = lane%16)
            float mk = mask[b * SEQ + ng];
#pragma unroll
            for (int r = 0; r < 8; ++r) {
                int qg = qt * 16 + off + r;
                sc[t][r] += s_bias[w][s_lut[ng - qg + (SEQ - 1)]] + mk;
            }
        }
#pragma unroll
        for (int r = 0; r < 8; ++r) {
            float v0 = fmaxf(sc[0][r], sc[1][r]);
            v0 = fmaxf(v0, __shfl_xor(v0, 1, 32));          // 16-lane (half) reduce
            v0 = fmaxf(v0, __shfl_xor(v0, 2, 32));
            v0 = fmaxf(v0, __shfl_xor(v0, 4, 32));
            v0 = fmaxf(v0, __shfl_xor(v0, 8, 32));
            float mnew  = fmaxf(rmax[r], v0);
            float scale = __expf(rmax[r] - mnew);
            rmax[r] = mnew;
            float p0 = __expf(sc[0][r] - mnew);
            float p1 = __expf(sc[1][r] - mnew);
            float ps = p0 + p1;
            ps += __shfl_xor(ps, 1, 32);
            ps += __shfl_xor(ps, 2, 32);
            ps += __shfl_xor(ps, 4, 32);
            ps += __shfl_xor(ps, 8, 32);
            rsum[r] = rsum[r] * scale + ps;
#pragma unroll
            for (int t = 0; t < 4; ++t) acc[t][r] *= scale;
            s_p[w][off + r][m]      = (__bf16)p0;           // C-layout -> LDS
            s_p[w][off + r][16 + m] = (__bf16)p1;
        }
        asm volatile("s_wait_dscnt 0" ::: "memory");        // same-wave LDS fence

        // P as A-fragment (contiguous 16B LDS reads in A layout)
        BF16Frag ap;
        ap.q[0] = *(const uint4*)&s_p[w][m][base];
        ap.q[1] = *(const uint4*)&s_p[w][m][16 + base];

        // acc += P(16x32) * V(32x64): V B-frag lane <-> d col, elements over keys
        const __bf16* vrow = V + (size_t)(b * SEQ + k0) * DM + h * DKV + m;
#pragma unroll
        for (int t = 0; t < 4; ++t) {
            BF16Frag bv;
#pragma unroll
            for (int j = 0; j < 16; ++j)
                bv.e[j] = vrow[(size_t)((j & 7) + base + ((j >> 3) << 4)) * DM + t * 16];
            acc[t] = wmma_bf16(ap.v, bv.v, acc[t]);
        }
    }

    // epilogue: context[b,s,h*64+d] = acc / rowsum  (bf16 for final WMMA GEMM)
#pragma unroll
    for (int t = 0; t < 4; ++t)
#pragma unroll
        for (int r = 0; r < 8; ++r)
            Ctx[(size_t)(b * SEQ + qt * 16 + off + r) * DM + h * DKV + t * 16 + m] =
                (__bf16)(acc[t][r] / rsum[r]);
}

// ---------------------------------------------------------------------------
// Kernel 5: out_f32[M,N] = A_bf16[M,K] @ B_f32[K,N]  (output projection)
// ---------------------------------------------------------------------------
__global__ void __launch_bounds__(128)
gemm_bf16_f32(const __bf16* __restrict__ A, const float* __restrict__ Bm,
              float* __restrict__ C, int M, int N, int K) {
    int lane = threadIdx.x & 31;
    int w    = threadIdx.x >> 5;
    int tile = blockIdx.x * 4 + w;
    int ntn  = N / 64;
    int row0 = (tile / ntn) * 16;
    int col0 = (tile % ntn) * 64;
    if (row0 >= M) return;

    int m    = lane & 15;
    int base = (lane < 16) ? 0 : 8;
    int off  = (lane < 16) ? 0 : 8;

    v8f_t acc[4] = {};
    for (int c = 0; c < K / 32; ++c) {
        BF16Frag a;
        const __bf16* arow = A + (size_t)(row0 + m) * K + c * 32;
        a.q[0] = *(const uint4*)(arow + base);
        a.q[1] = *(const uint4*)(arow + 16 + base);

        const float* brow = Bm + (size_t)(c * 32) * N + col0 + m;
#pragma unroll
        for (int t = 0; t < 4; ++t) {
            BF16Frag b;
#pragma unroll
            for (int j = 0; j < 16; ++j) b.e[j] = (__bf16)brow[(size_t)((j & 7) + base + ((j >> 3) << 4)) * N + t * 16];
            acc[t] = wmma_bf16(a.v, b.v, acc[t]);
        }
    }
#pragma unroll
    for (int t = 0; t < 4; ++t)
#pragma unroll
        for (int r = 0; r < 8; ++r)
            C[(size_t)(row0 + off + r) * N + col0 + t * 16 + m] = acc[t][r];
}

// ---------------------------------------------------------------------------
extern "C" void kernel_launch(void* const* d_in, const int* in_sizes, int n_in,
                              void* d_out, int out_size, void* d_ws, size_t ws_size,
                              hipStream_t stream) {
    (void)in_sizes; (void)n_in; (void)out_size; (void)ws_size;
    const float* X    = (const float*)d_in[0];   // [2,2048,1024]
    const float* mask = (const float*)d_in[1];   // [2,1,1,2048]
    const float* Wq   = (const float*)d_in[2];   // [1024,1024]
    const float* Wk   = (const float*)d_in[3];
    const float* Wv   = (const float*)d_in[4];
    const float* Wo   = (const float*)d_in[5];
    const float* rel  = (const float*)d_in[6];   // [32,16]

    float* out      = (float*)d_out;                         // [2,2048,1024]
    float* bias_out = out + (size_t)BATCH * SEQ * DM;        // [1,16,2048,2048]

    const int    M    = BATCH * SEQ;                         // 4096
    const size_t MATB = (size_t)M * DM;                      // 4,194,304 elems
    __bf16* Qb  = (__bf16*)d_ws;
    __bf16* Kb  = Qb + MATB;
    __bf16* Vb  = Kb + MATB;
    __bf16* Ctx = Vb + MATB;
    int*    lut = (int*)(Ctx + MATB);                        // ~33.6 MB total ws

    t5_bucket_lut<<<(LUT_N + 255) / 256, 256, 0, stream>>>(lut);
    t5_bias_out<<<HEADS * SEQ, 256, 0, stream>>>(lut, rel, bias_out);

    const int gemm_blocks = (M / 16) * (DM / 64) / 4;        // 1024
    gemm_f32_bf16<<<gemm_blocks, 128, 0, stream>>>(X, Wq, Qb, M, DM, DM);
    gemm_f32_bf16<<<gemm_blocks, 128, 0, stream>>>(X, Wk, Kb, M, DM, DM);
    gemm_f32_bf16<<<gemm_blocks, 128, 0, stream>>>(X, Wv, Vb, M, DM, DM);

    const int attn_blocks = BATCH * HEADS * (SEQ / 16) / 4;  // 1024
    t5_flash_attn<<<attn_blocks, 128, 0, stream>>>(Qb, Kb, Vb, mask, rel, lut, Ctx);

    gemm_bf16_f32<<<gemm_blocks, 128, 0, stream>>>(Ctx, Wo, out, M, DM, DM);
}
